// PCT_49804440764996
// MI455X (gfx1250) — matrix operand order repaired and checked by reading, not compiled, not run on hardware
//
#include <hip/hip_runtime.h>

// ---------------------------------------------------------------------------
// PCT (Point Cloud Transformer) forward for MI455X / gfx1250 (wave32, WMMA).
// All GEMM-like work lowered to v_wmma_f32_16x16x32_f16 (f16 in, f32 acc).
// Fast GEMM: 4-wave blocks, 64x64 macro-tiles, B panel staged in LDS,
// b128 fragment loads, global_prefetch for the next k panel.
// ---------------------------------------------------------------------------

typedef __attribute__((ext_vector_type(16))) _Float16 v16h;
typedef __attribute__((ext_vector_type(8)))  _Float16 v8h;
typedef __attribute__((ext_vector_type(4)))  _Float16 v4h;
typedef __attribute__((ext_vector_type(8)))  float    v8f;

__device__ __forceinline__ v8f wmma32(v16h a, v16h b, v8f c) {
  return __builtin_amdgcn_wmma_f32_16x16x32_f16(false, a, false, b, (short)0, c,
                                                false, false);
}

// ---- fast (k-contiguous) fragment loaders: two b128 loads per fragment ----
// A (16x32, f16): lane L = row M=L&15; lanes<16 hold K 0..7 / 16..23,
// lanes>=16 hold K 8..15 / 24..31 -> contiguous runs of 8 halves.
__device__ __forceinline__ v16h frag_a_fast(const _Float16* p0, int rs) {
  int lane = threadIdx.x & 31;
  const _Float16* p = p0 + (lane & 15) * rs + ((lane & 16) ? 8 : 0);
  v8h lo = *(const v8h*)p;
  v8h hi = *(const v8h*)(p + 16);
  v16h f;
#pragma unroll
  for (int i = 0; i < 8; ++i) { f[i] = lo[i]; f[8 + i] = hi[i]; }
  return f;
}
// B (32x16, f16): lane L = col N=L&15; lanes<16 hold K 0..7 / 16..23,
// lanes>=16 hold K 8..15 / 24..31 -> contiguous runs of 8 halves.
__device__ __forceinline__ v16h frag_b_fast(const _Float16* p0, int cs) {
  int lane = threadIdx.x & 31;
  const _Float16* p = p0 + (lane & 15) * cs + ((lane & 16) ? 8 : 0);
  v8h lo = *(const v8h*)p;
  v8h hi = *(const v8h*)(p + 16);
  v16h f;
#pragma unroll
  for (int i = 0; i < 8; ++i) { f[i] = lo[i]; f[8 + i] = hi[i]; }
  return f;
}

// ---- generic strided loaders (used only by the att^T x V GEMM) ----
__device__ __forceinline__ v16h frag_a(const _Float16* p0, int rs, int ks) {
  int lane = threadIdx.x & 31;
  const _Float16* p = p0 + (lane & 15) * rs;
  int ko = (lane & 16) ? 8 : 0;
  v16h f;
#pragma unroll
  for (int i = 0; i < 4; ++i) {
    f[2 * i]     = p[(ko + 2 * i) * ks];
    f[2 * i + 1] = p[(ko + 2 * i + 1) * ks];
    f[8 + 2 * i]     = p[(16 + ko + 2 * i) * ks];
    f[8 + 2 * i + 1] = p[(16 + ko + 2 * i + 1) * ks];
  }
  return f;
}
__device__ __forceinline__ v16h frag_b(const _Float16* p0, int cs, int ks) {
  int lane = threadIdx.x & 31;
  const _Float16* p = p0 + (lane & 15) * cs;
  int ko = (lane & 16) ? 8 : 0;
  v16h f;
#pragma unroll
  for (int i = 0; i < 8; ++i) { f[i] = p[(ko + i) * ks]; f[8 + i] = p[(16 + ko + i) * ks]; }
  return f;
}

// D/C f32 16x16: lane = N-col (L&15); VGPR r = M row r (+8 for lanes>=16).
__device__ __forceinline__ void store_d(float* p0, int rs, const v8f& d) {
  int lane = threadIdx.x & 31;
  int mo = (lane & 16) ? 8 : 0;
  float* p = p0 + (lane & 15);
#pragma unroll
  for (int r = 0; r < 8; ++r) p[(long)(mo + r) * rs] = d[r];
}

// ---------------------------------------------------------------------------
// Fast batched GEMM (aks==1, bks==1): C[b](MxN,f32) = A[b](MxK) * B[b](KxN)
// block = 128 threads (4 waves); grid = (N/64, ceil(M/64), batches).
// Each wave owns one 16-row m-tile and 64 columns; the 64x32 B panel for the
// current k-step is staged cooperatively into LDS and shared by all 4 waves.
// ---------------------------------------------------------------------------
__global__ __launch_bounds__(128) void k_gemm4(const _Float16* __restrict__ A,
                                               const _Float16* __restrict__ Bm,
                                               float* __restrict__ C, int M,
                                               int N, int K, long abs_, long bbs,
                                               long cbs, int ars, int bcs) {
  __shared__ _Float16 Bt[64 * 32];
  int wave = threadIdx.x >> 5, lane = threadIdx.x & 31;
  int n0 = blockIdx.x * 64;
  int m0 = blockIdx.y * 64 + wave * 16;
  int b = blockIdx.z;
  bool active = (m0 + 16 <= M);
  const _Float16* Ab = A + (long)b * abs_ + (long)m0 * ars;
  const _Float16* Bb = Bm + (long)b * bbs + (long)n0 * bcs;
  int scol = threadIdx.x >> 1;
  int sho = (threadIdx.x & 1) * 16;
  v8f acc[4] = {};
  for (int k = 0; k < K; k += 32) {
    // cooperative stage of the 64x32 B panel (each thread: 2 x b128)
    {
      const _Float16* src = Bb + (long)scol * bcs + k + sho;
      _Float16* dst = Bt + scol * 32 + sho;
      *(v8h*)dst = *(const v8h*)src;
      *(v8h*)(dst + 8) = *(const v8h*)(src + 8);
      if (k + 32 < K) {
        __builtin_prefetch(src + 32, 0, 1);   // global_prefetch_b8 (next panel)
        if (active) __builtin_prefetch(Ab + k + 32 + (long)lane * ars, 0, 1);
      }
    }
    __syncthreads();
    if (active) {
      v16h a = frag_a_fast(Ab + k, ars);
#pragma unroll
      for (int t = 0; t < 4; ++t)
        acc[t] = wmma32(a, frag_b_fast(Bt + t * 16 * 32, 32), acc[t]);
    }
    __syncthreads();
  }
  if (active) {
    float* Cb = C + (long)b * cbs + (long)m0 * N + n0;
#pragma unroll
    for (int t = 0; t < 4; ++t) store_d(Cb + t * 16, N, acc[t]);
  }
}

// Generic strided GEMM (one wave / 16x16 tile) — only for XR = att^T * V.
__global__ __launch_bounds__(32) void k_gemm_s(const _Float16* __restrict__ A,
                                               const _Float16* __restrict__ Bm,
                                               float* __restrict__ C, int N,
                                               int K, long abs_, long bbs,
                                               long cbs, int ars, int aks,
                                               int bcs, int bks) {
  int n0 = blockIdx.x * 16, m0 = blockIdx.y * 16, b = blockIdx.z;
  const _Float16* Ab = A + (long)b * abs_ + (long)m0 * ars;
  const _Float16* Bb = Bm + (long)b * bbs + (long)n0 * bcs;
  v8f acc = {};
  for (int k = 0; k < K; k += 32)
    acc = wmma32(frag_a(Ab + (long)k * aks, ars, aks),
                 frag_b(Bb + (long)k * bks, bcs, bks), acc);
  store_d(C + (long)b * cbs + (long)m0 * N + n0, N, acc);
}

// ---------------------------------------------------------------------------
// BN param prep: scale = g/sqrt(v+eps), shift = b - m*scale
// ---------------------------------------------------------------------------
__global__ void k_bn_prep(const float* g, const float* b, const float* m,
                          const float* v, float* sc, float* sh, int C) {
  int i = blockIdx.x * blockDim.x + threadIdx.x;
  if (i >= C) return;
  float s = g[i] * rsqrtf(v[i] + 1e-5f);
  sc[i] = s;
  sh[i] = b[i] - m[i] * s;
}

// ---------------------------------------------------------------------------
// Generic pointwise epilogue (channel-fastest layout, channel = i % C):
//   v = x[i] (+bias[c]) ; bn ; act(0 none,1 relu,2 lrelu0.2) ; (+residual[i])
//   optional f32 store, optional strided f16 store (row = i/C).
// Also plain f32->f16 converter (everything null, C=1).
// ---------------------------------------------------------------------------
__global__ void k_epilogue(const float* __restrict__ x, const float* bias,
                           const float* scale, const float* shift,
                           const float* residual, int C, long total, int act,
                           float* out32, _Float16* out16, int o16s) {
  long i = (long)blockIdx.x * blockDim.x + threadIdx.x;
  if (i >= total) return;
  int c = (int)(i % C);
  float v = x[i];
  if (bias) v += bias[c];
  if (scale) v = v * scale[c] + shift[c];
  if (act == 1) v = fmaxf(v, 0.f);
  else if (act == 2) v = (v > 0.f) ? v : 0.2f * v;
  if (residual) v += residual[i];
  if (out32) out32[i] = v;
  if (out16) out16[(i / C) * (long)o16s + c] = (_Float16)v;
}

// conv1 (3->64) + BN + ReLU -> f16 [B][N][64]; also emit xyz [B][N][3]
__global__ void k_conv1(const float* __restrict__ x, const float* __restrict__ w,
                        const float* sc, const float* sh,
                        _Float16* __restrict__ h1h, float* __restrict__ xyz,
                        int N) {
  long i = (long)blockIdx.x * blockDim.x + threadIdx.x;
  if (i >= (long)32 * N * 64) return;
  int o = (int)(i & 63);
  long bn_ = i >> 6;
  int n = (int)(bn_ % N);
  int b = (int)(bn_ / N);
  const float* xb = x + (long)b * 3 * N + n;
  float a = w[o * 3 + 0] * xb[0] + w[o * 3 + 1] * xb[N] + w[o * 3 + 2] * xb[2 * N];
  a = fmaxf(a * sc[o] + sh[o], 0.f);
  h1h[i] = (_Float16)a;
  if (o < 3) xyz[bn_ * 3 + o] = xb[(long)o * N];
}

// ---------------------------------------------------------------------------
// Farthest point sampling: one 256-thread block per batch, dists in LDS.
// ---------------------------------------------------------------------------
__global__ __launch_bounds__(256) void k_fps(const float* __restrict__ xyz,
                                             int* __restrict__ out, int N, int S) {
  __shared__ float dists[2048];
  __shared__ float rv[256];
  __shared__ int ri[256];
  __shared__ int s_far;
  int b = blockIdx.x, t = threadIdx.x;
  for (int i = t; i < N; i += 256) dists[i] = 1e10f;
  if (t == 0) s_far = 0;
  __syncthreads();
  for (int it = 0; it < S; ++it) {
    int far = s_far;
    if (t == 0) out[b * S + it] = far;
    const float* fp = xyz + ((long)b * N + far) * 3;
    float fx = fp[0], fy = fp[1], fz = fp[2];
    float bmax = -1.f; int bidx = 0;
    for (int i = t; i < N; i += 256) {
      const float* p = xyz + ((long)b * N + i) * 3;
      float dx = p[0] - fx, dy = p[1] - fy, dz = p[2] - fz;
      float nd = fminf(dists[i], dx * dx + dy * dy + dz * dz);
      dists[i] = nd;
      if (nd > bmax) { bmax = nd; bidx = i; }
    }
    rv[t] = bmax; ri[t] = bidx;
    __syncthreads();
    for (int off = 128; off > 0; off >>= 1) {
      if (t < off && rv[t + off] > rv[t]) { rv[t] = rv[t + off]; ri[t] = ri[t + off]; }
      __syncthreads();
    }
    if (t == 0) s_far = ri[0];
    __syncthreads();
  }
}

// gather sampled xyz: out[b][s] = xyz[b][fidx[b][s]]
__global__ void k_gather3(const float* __restrict__ xyz, const int* __restrict__ fidx,
                          float* __restrict__ out, int Nsrc, int S) {
  int i = blockIdx.x * blockDim.x + threadIdx.x;
  if (i >= 32 * S) return;
  int b = i / S, s = i % S;
  int id = fidx[b * S + s];
  const float* p = xyz + ((long)b * Nsrc + id) * 3;
  float* q = out + (long)i * 3;
  q[0] = p[0]; q[1] = p[1]; q[2] = p[2];
}

// ---------------------------------------------------------------------------
// kNN (32 nearest of N): one wave per (b,s); per-lane PPL distances in regs,
// 32 rounds of masked wave-argmin via __shfl_xor.
// ---------------------------------------------------------------------------
template <int PPL>
__global__ __launch_bounds__(32) void k_knn(const float* __restrict__ xyz,
                                            const float* __restrict__ nxyz,
                                            int* __restrict__ knn, int N, int S) {
  int s = blockIdx.x, b = blockIdx.y, lane = threadIdx.x;
  const float* c = nxyz + ((long)b * S + s) * 3;
  float cx = c[0], cy = c[1], cz = c[2];
  float d[PPL];
#pragma unroll
  for (int j = 0; j < PPL; ++j) {
    const float* p = xyz + ((long)b * N + lane * PPL + j) * 3;
    float dx = p[0] - cx, dy = p[1] - cy, dz = p[2] - cz;
    d[j] = dx * dx + dy * dy + dz * dz;
  }
  unsigned long long taken = 0ull;
  int mywin = 0;
  for (int r = 0; r < 32; ++r) {
    float bv = 3e38f; int bj = -1;
#pragma unroll
    for (int j = 0; j < PPL; ++j)
      if (!((taken >> j) & 1ull) && d[j] < bv) { bv = d[j]; bj = j; }
    int bi = (bj >= 0) ? (lane * PPL + bj) : 0x7fffffff;
    for (int off = 16; off > 0; off >>= 1) {
      float ov = __shfl_xor(bv, off, 32);
      int   oi = __shfl_xor(bi, off, 32);
      if (ov < bv || (ov == bv && oi < bi)) { bv = ov; bi = oi; }
    }
    if (bi >= lane * PPL && bi < (lane + 1) * PPL) taken |= 1ull << (bi - lane * PPL);
    if (r == lane) mywin = bi;
  }
  knn[((long)(b * S + s)) * 32 + lane] = mywin;
}

// ---------------------------------------------------------------------------
// Fused local encoder: gather 32 neighbors (rel|center) -> [32, 2D] f16 tile
// in LDS (float4 loads, v4h stores), WMMA GEMM (2D->O) + BN + ReLU -> LDS,
// WMMA GEMM (O->O) + BN + ReLU, max over k, write [B][S][O] f32.
// One wave per (b,s).
// ---------------------------------------------------------------------------
template <int D, int O>
__global__ __launch_bounds__(32) void k_local_enc(
    const float* __restrict__ pts, int Npts, const int* __restrict__ fidx,
    const int* __restrict__ knn, int S, const _Float16* __restrict__ w1,
    const float* sc1, const float* sh1, const _Float16* __restrict__ w2,
    const float* sc2, const float* sh2, float* __restrict__ out) {
  constexpr int D2 = 2 * D;
  __shared__ _Float16 At[32 * D2];
  __shared__ _Float16 Ht[32 * O];
  int s = blockIdx.x, b = blockIdx.y, lane = threadIdx.x;
  {
    int ni = knn[((long)(b * S + s)) * 32 + lane];
    int ci = fidx[b * S + s];
    const float4* gp = (const float4*)(pts + ((long)b * Npts + ni) * D);
    const float4* cp = (const float4*)(pts + ((long)b * Npts + ci) * D);
#pragma unroll
    for (int c4 = 0; c4 < D / 4; ++c4) {
      float4 g = gp[c4], cf = cp[c4];
      v4h rel, cen;
      rel[0] = (_Float16)(g.x - cf.x); rel[1] = (_Float16)(g.y - cf.y);
      rel[2] = (_Float16)(g.z - cf.z); rel[3] = (_Float16)(g.w - cf.w);
      cen[0] = (_Float16)cf.x; cen[1] = (_Float16)cf.y;
      cen[2] = (_Float16)cf.z; cen[3] = (_Float16)cf.w;
      *(v4h*)&At[lane * D2 + 4 * c4] = rel;
      *(v4h*)&At[lane * D2 + D + 4 * c4] = cen;
    }
  }
  __syncthreads();
  int ch15 = lane & 15;
  int mo = (lane & 16) ? 8 : 0;
  for (int nt = 0; nt < O / 16; ++nt) {
    float sc = sc1[nt * 16 + ch15], sh = sh1[nt * 16 + ch15];
    for (int mt = 0; mt < 2; ++mt) {
      v8f acc = {};
      for (int kt = 0; kt < D2 / 32; ++kt)
        acc = wmma32(frag_a_fast(At + mt * 16 * D2 + kt * 32, D2),
                     frag_b_fast(w1 + (long)(nt * 16) * D2 + kt * 32, D2), acc);
#pragma unroll
      for (int r = 0; r < 8; ++r) {
        float hv = fmaxf(acc[r] * sc + sh, 0.f);
        Ht[(mt * 16 + mo + r) * O + nt * 16 + ch15] = (_Float16)hv;
      }
    }
  }
  __syncthreads();
  for (int nt = 0; nt < O / 16; ++nt) {
    float sc = sc2[nt * 16 + ch15], sh = sh2[nt * 16 + ch15];
    float cmax = -3e38f;
    for (int mt = 0; mt < 2; ++mt) {
      v8f acc = {};
      for (int kt = 0; kt < O / 32; ++kt)
        acc = wmma32(frag_a_fast(Ht + mt * 16 * O + kt * 32, O),
                     frag_b_fast(w2 + (long)(nt * 16) * O + kt * 32, O), acc);
#pragma unroll
      for (int r = 0; r < 8; ++r)
        cmax = fmaxf(cmax, fmaxf(acc[r] * sc + sh, 0.f));
    }
    cmax = fmaxf(cmax, __shfl_xor(cmax, 16, 32));
    if (lane < 16) out[((long)(b * S + s)) * O + nt * 16 + lane] = cmax;
  }
}

// softmax over rows of 256 (in place), one block per row
__global__ __launch_bounds__(256) void k_softmax(float* __restrict__ e) {
  __shared__ float red[256];
  float* p = e + (long)blockIdx.x * 256;
  int t = threadIdx.x;
  float v = p[t];
  red[t] = v;
  __syncthreads();
  for (int off = 128; off > 0; off >>= 1) {
    if (t < off) red[t] = fmaxf(red[t], red[t + off]);
    __syncthreads();
  }
  float mx = red[0];
  __syncthreads();
  float ev = __expf(v - mx);
  red[t] = ev;
  __syncthreads();
  for (int off = 128; off > 0; off >>= 1) {
    if (t < off) red[t] += red[t + off];
    __syncthreads();
  }
  p[t] = ev / red[0];
}

// column sums of att [B][256][256] over n -> cs [B][256]
__global__ void k_colsum(const float* __restrict__ att, float* __restrict__ cs) {
  int i = blockIdx.x * blockDim.x + threadIdx.x;
  if (i >= 32 * 256) return;
  int b = i / 256, m = i % 256;
  const float* p = att + (long)b * 256 * 256 + m;
  float s = 0.f;
  for (int n = 0; n < 256; ++n) s += p[(long)n * 256];
  cs[i] = s;
}

__global__ void k_colnorm(const float* __restrict__ att, const float* __restrict__ cs,
                          _Float16* __restrict__ out, long total) {
  long i = (long)blockIdx.x * blockDim.x + threadIdx.x;
  if (i >= total) return;
  int b = (int)(i >> 16);           // / (256*256)
  int m = (int)(i & 255);
  out[i] = (_Float16)(att[i] / (1e-9f + cs[b * 256 + m]));
}

__global__ void k_sub16(const float* __restrict__ a, const float* __restrict__ b,
                        _Float16* __restrict__ o, long total) {
  long i = (long)blockIdx.x * blockDim.x + threadIdx.x;
  if (i < total) o[i] = (_Float16)(a[i] - b[i]);
}

// global max over the 256 points: F [B][256][512] -> g [B][512]
__global__ void k_maxn(const float* __restrict__ F, float* __restrict__ g) {
  int i = blockIdx.x * blockDim.x + threadIdx.x;
  if (i >= 32 * 512) return;
  int b = i / 512, c = i % 512;
  const float* p = F + (long)b * 256 * 512 + c;
  float m = -3e38f;
  for (int n = 0; n < 256; ++n) m = fmaxf(m, p[(long)n * 512]);
  g[i] = m;
}

// final linear 256->40
__global__ void k_lin3(const float* __restrict__ g, const float* __restrict__ w,
                       const float* __restrict__ bias, float* __restrict__ out) {
  int i = blockIdx.x * blockDim.x + threadIdx.x;
  if (i >= 32 * 40) return;
  int b = i / 40, o = i % 40;
  const float* gv = g + b * 256;
  const float* wv = w + o * 256;
  float acc = bias[o];
  for (int c = 0; c < 256; ++c) acc += gv[c] * wv[c];
  out[i] = acc;
}

// ---------------------------------------------------------------------------
// Host orchestration
// ---------------------------------------------------------------------------
// d_in flattening (insertion order of setup_inputs):
enum {
  IX_X = 0, IX_C1W = 1, IX_BN1 = 2, IX_C2W = 6, IX_BN2 = 7,
  IX_LE0C1 = 11, IX_LE0B1 = 12, IX_LE0C2 = 16, IX_LE0B2 = 17,
  IX_LE1C1 = 21, IX_LE1B1 = 22, IX_LE1C2 = 26, IX_LE1B2 = 27,
  IX_PTC1 = 31, IX_PTB1 = 32, IX_PTC2 = 36, IX_PTB2 = 37,
  IX_SA0 = 41,  // per layer: qk, v_w, v_b, t_w, t_b, bn{g,b,m,v} (9 entries)
  IX_FUSEW = 77, IX_FUSEBN = 78, IX_LIN1 = 82, IX_BN6 = 83,
  IX_LIN2 = 87, IX_LIN2B = 88, IX_BN7 = 89, IX_LIN3 = 93, IX_LIN3B = 94
};

extern "C" void kernel_launch(void* const* d_in, const int* in_sizes, int n_in,
                              void* d_out, int out_size, void* d_ws, size_t ws_size,
                              hipStream_t stream) {
  (void)in_sizes; (void)n_in; (void)out_size; (void)ws_size;
  const int B = 32, N = 2048;
  auto F = [&](int i) { return (const float*)d_in[i]; };

  char* base = (char*)d_ws;
  size_t off = 0;
  auto alloc = [&](size_t bytes) -> void* {
    void* p = base + off;
    off += (bytes + 255) & ~(size_t)255;
    return p;
  };

  // --- BN scale/shift prep ---
  auto bnprep = [&](int gi, int C, float** sc, float** sh) {
    *sc = (float*)alloc((size_t)C * 4);
    *sh = (float*)alloc((size_t)C * 4);
    k_bn_prep<<<(C + 63) / 64, 64, 0, stream>>>(F(gi), F(gi + 1), F(gi + 2),
                                                F(gi + 3), *sc, *sh, C);
  };
  float *sc1, *sh1, *sc2, *sh2;
  bnprep(IX_BN1, 64, &sc1, &sh1);
  bnprep(IX_BN2, 64, &sc2, &sh2);
  float *l0s1, *l0h1, *l0s2, *l0h2, *l1s1, *l1h1, *l1s2, *l1h2;
  bnprep(IX_LE0B1, 128, &l0s1, &l0h1);
  bnprep(IX_LE0B2, 128, &l0s2, &l0h2);
  bnprep(IX_LE1B1, 256, &l1s1, &l1h1);
  bnprep(IX_LE1B2, 256, &l1s2, &l1h2);
  float *p1s, *p1h, *p2s, *p2h;
  bnprep(IX_PTB1, 256, &p1s, &p1h);
  bnprep(IX_PTB2, 256, &p2s, &p2h);
  float *sas[4], *sah[4];
  for (int j = 0; j < 4; ++j) bnprep(IX_SA0 + 9 * j + 5, 256, &sas[j], &sah[j]);
  float *fus, *fuh, *b6s, *b6h, *b7s, *b7h;
  bnprep(IX_FUSEBN, 512, &fus, &fuh);
  bnprep(IX_BN6, 512, &b6s, &b6h);
  bnprep(IX_BN7, 256, &b7s, &b7h);

  // --- weight f32 -> f16 ---
  auto tohalf = [&](int gi, long n) -> _Float16* {
    _Float16* w = (_Float16*)alloc((size_t)n * 2);
    k_epilogue<<<(int)((n + 255) / 256), 256, 0, stream>>>(
        F(gi), nullptr, nullptr, nullptr, nullptr, 1, n, 0, nullptr, w, 1);
    return w;
  };
  _Float16* c2h = tohalf(IX_C2W, 64 * 64);
  _Float16* le0c1h = tohalf(IX_LE0C1, 128 * 128);
  _Float16* le0c2h = tohalf(IX_LE0C2, 128 * 128);
  _Float16* le1c1h = tohalf(IX_LE1C1, 256 * 256);
  _Float16* le1c2h = tohalf(IX_LE1C2, 256 * 256);
  _Float16* ptc1h = tohalf(IX_PTC1, 256 * 256);
  _Float16* ptc2h = tohalf(IX_PTC2, 256 * 256);
  _Float16 *qkh[4], *vwh[4], *twh[4];
  for (int j = 0; j < 4; ++j) {
    qkh[j] = tohalf(IX_SA0 + 9 * j + 0, 64 * 256);
    vwh[j] = tohalf(IX_SA0 + 9 * j + 1, 256 * 256);
    twh[j] = tohalf(IX_SA0 + 9 * j + 3, 256 * 256);
  }
  _Float16* fuseh = tohalf(IX_FUSEW, 512 * 1280);
  _Float16* lin1h = tohalf(IX_LIN1, 512 * 512);
  _Float16* lin2h = tohalf(IX_LIN2, 256 * 512);

  // --- activations / scratch ---
  _Float16* h1h = (_Float16*)alloc((size_t)B * N * 64 * 2);
  float* pts32 = (float*)alloc((size_t)B * N * 64 * 4);
  float* xyz = (float*)alloc((size_t)B * N * 3 * 4);
  int* fps1 = (int*)alloc((size_t)B * 512 * 4);
  float* nxyz1 = (float*)alloc((size_t)B * 512 * 3 * 4);
  int* knn1 = (int*)alloc((size_t)B * 512 * 32 * 4);
  float* f0 = (float*)alloc((size_t)B * 512 * 128 * 4);
  int* fps2 = (int*)alloc((size_t)B * 256 * 4);
  float* nxyz2 = (float*)alloc((size_t)B * 256 * 3 * 4);
  int* knn2 = (int*)alloc((size_t)B * 256 * 32 * 4);
  float* f1_32 = (float*)alloc((size_t)B * 256 * 256 * 4);
  _Float16* xcat = (_Float16*)alloc((size_t)B * 256 * 1280 * 2);
  _Float16* Xh = (_Float16*)alloc((size_t)B * 256 * 256 * 2);
  float* X32 = (float*)alloc((size_t)B * 256 * 256 * 4);
  float* tmp32 = (float*)alloc((size_t)B * 256 * 256 * 4);
  float* Q32 = (float*)alloc((size_t)B * 256 * 64 * 4);
  _Float16* Qh = (_Float16*)alloc((size_t)B * 256 * 64 * 2);
  float* E32 = (float*)alloc((size_t)B * 256 * 256 * 4);
  float* cs = (float*)alloc((size_t)B * 256 * 4);
  _Float16* att_h = (_Float16*)alloc((size_t)B * 256 * 256 * 2);
  float* V32 = (float*)alloc((size_t)B * 256 * 256 * 4);
  _Float16* Vh = (_Float16*)alloc((size_t)B * 256 * 256 * 2);
  float* XR32 = (float*)alloc((size_t)B * 256 * 256 * 4);
  _Float16* Sh = (_Float16*)alloc((size_t)B * 256 * 256 * 2);
  float* Fz = (float*)alloc((size_t)B * 256 * 512 * 4);
  float* g32 = (float*)alloc((size_t)B * 512 * 4);
  _Float16* gh = (_Float16*)alloc((size_t)B * 512 * 2);
  float* L1 = (float*)alloc((size_t)B * 512 * 4);
  _Float16* g1h = (_Float16*)alloc((size_t)B * 512 * 2);
  float* L2 = (float*)alloc((size_t)B * 256 * 4);
  float* g2 = (float*)alloc((size_t)B * 256 * 4);

  long tot;

  // --- stem: conv1 (VALU) + conv2 (WMMA) ---
  tot = (long)B * N * 64;
  k_conv1<<<(int)((tot + 255) / 256), 256, 0, stream>>>(F(IX_X), F(IX_C1W), sc1,
                                                        sh1, h1h, xyz, N);
  k_gemm4<<<dim3(1, N / 64, B), 128, 0, stream>>>(h1h, c2h, pts32, N, 64, 64,
                                                  (long)N * 64, 0, (long)N * 64,
                                                  64, 64);
  k_epilogue<<<(int)((tot + 255) / 256), 256, 0, stream>>>(
      pts32, nullptr, sc2, sh2, nullptr, 64, tot, 1, pts32, nullptr, 1);

  // --- stage 1: FPS 512, kNN 32 (of 2048), local encoder 128->128 ---
  k_fps<<<B, 256, 0, stream>>>(xyz, fps1, N, 512);
  k_gather3<<<(B * 512 + 255) / 256, 256, 0, stream>>>(xyz, fps1, nxyz1, N, 512);
  k_knn<64><<<dim3(512, B), 32, 0, stream>>>(xyz, nxyz1, knn1, N, 512);
  k_local_enc<64, 128><<<dim3(512, B), 32, 0, stream>>>(
      pts32, N, fps1, knn1, 512, le0c1h, l0s1, l0h1, le0c2h, l0s2, l0h2, f0);

  // --- stage 2: FPS 256 of 512, kNN 32, local encoder 256->256 ---
  k_fps<<<B, 256, 0, stream>>>(nxyz1, fps2, 512, 256);
  k_gather3<<<(B * 256 + 255) / 256, 256, 0, stream>>>(nxyz1, fps2, nxyz2, 512, 256);
  k_knn<16><<<dim3(256, B), 32, 0, stream>>>(nxyz1, nxyz2, knn2, 512, 256);
  k_local_enc<128, 256><<<dim3(256, B), 32, 0, stream>>>(
      f0, 512, fps2, knn2, 256, le1c1h, l1s1, l1h1, le1c2h, l1s2, l1h2, f1_32);

  // f1 -> xcat slice [1024..1280) (f16)
  tot = (long)B * 256 * 256;
  int blk = (int)((tot + 255) / 256);
  k_epilogue<<<blk, 256, 0, stream>>>(f1_32, nullptr, nullptr, nullptr, nullptr,
                                      256, tot, 0, nullptr, xcat + 1024, 1280);

  // --- PT convs ---
  k_gemm4<<<dim3(4, 4, B), 128, 0, stream>>>(xcat + 1024, ptc1h, tmp32, 256, 256,
                                             256, 256L * 1280, 0, 256L * 256,
                                             1280, 256);
  k_epilogue<<<blk, 256, 0, stream>>>(tmp32, nullptr, p1s, p1h, nullptr, 256, tot,
                                      1, nullptr, Xh, 256);
  k_gemm4<<<dim3(4, 4, B), 128, 0, stream>>>(Xh, ptc2h, tmp32, 256, 256, 256,
                                             256L * 256, 0, 256L * 256, 256, 256);
  k_epilogue<<<blk, 256, 0, stream>>>(tmp32, nullptr, p2s, p2h, nullptr, 256, tot,
                                      1, X32, Xh, 256);

  // --- 4 offset-attention layers ---
  const _Float16* h16 = Xh;
  int h_rs = 256;
  long h_bs = 256L * 256;
  for (int j = 0; j < 4; ++j) {
    const float* v_b = F(IX_SA0 + 9 * j + 2);
    const float* t_b = F(IX_SA0 + 9 * j + 4);
    long qtot = (long)B * 256 * 64;
    // Q = X * qk^T  (shared weight => K matrix == Q matrix)
    k_gemm4<<<dim3(1, 4, B), 128, 0, stream>>>(h16, qkh[j], Q32, 256, 64, 256,
                                               h_bs, 0, 256L * 64, h_rs, 256);
    k_epilogue<<<(int)((qtot + 255) / 256), 256, 0, stream>>>(
        Q32, nullptr, nullptr, nullptr, nullptr, 1, qtot, 0, nullptr, Qh, 1);
    // energy = Q * Q^T
    k_gemm4<<<dim3(4, 4, B), 128, 0, stream>>>(Qh, Qh, E32, 256, 256, 64,
                                               256L * 64, 256L * 64, 256L * 256,
                                               64, 64);
    k_softmax<<<B * 256, 256, 0, stream>>>(E32);
    k_colsum<<<(B * 256 + 255) / 256, 256, 0, stream>>>(E32, cs);
    k_colnorm<<<blk, 256, 0, stream>>>(E32, cs, att_h, tot);
    // V = X * v_w^T + v_b
    k_gemm4<<<dim3(4, 4, B), 128, 0, stream>>>(h16, vwh[j], V32, 256, 256, 256,
                                               h_bs, 0, 256L * 256, h_rs, 256);
    k_epilogue<<<blk, 256, 0, stream>>>(V32, v_b, nullptr, nullptr, nullptr, 256,
                                        tot, 0, nullptr, Vh, 256);
    // XR = att^T * V   (A strided-transposed, B column-strided -> generic)
    k_gemm_s<<<dim3(16, 16, B), 32, 0, stream>>>(att_h, Vh, XR32, 256, 256,
                                                 256L * 256, 256L * 256,
                                                 256L * 256, 1, 256, 1, 256);
    // S = X - XR
    k_sub16<<<blk, 256, 0, stream>>>(X32, XR32, Sh, tot);
    // T = S * t_w^T ; out = X + relu(bn(T + t_b)) -> X32 and xcat slice j
    k_gemm4<<<dim3(4, 4, B), 128, 0, stream>>>(Sh, twh[j], tmp32, 256, 256, 256,
                                               256L * 256, 0, 256L * 256, 256,
                                               256);
    k_epilogue<<<blk, 256, 0, stream>>>(tmp32, t_b, sas[j], sah[j], X32, 256, tot,
                                        1, X32, xcat + (long)j * 256, 1280);
    h16 = xcat + (long)j * 256;
    h_rs = 1280;
    h_bs = 256L * 1280;
  }

  // --- fuse 1280 -> 512, BN, leaky relu, global max ---
  long ftot = (long)B * 256 * 512;
  k_gemm4<<<dim3(8, 4, B), 128, 0, stream>>>(xcat, fuseh, Fz, 256, 512, 1280,
                                             256L * 1280, 0, 256L * 512, 1280,
                                             1280);
  k_epilogue<<<(int)((ftot + 255) / 256), 256, 0, stream>>>(
      Fz, nullptr, fus, fuh, nullptr, 512, ftot, 2, Fz, nullptr, 1);
  k_maxn<<<(B * 512 + 255) / 256, 256, 0, stream>>>(Fz, g32);
  k_epilogue<<<(B * 512 + 255) / 256, 256, 0, stream>>>(
      g32, nullptr, nullptr, nullptr, nullptr, 1, (long)B * 512, 0, nullptr, gh, 1);

  // --- head: lin1 (WMMA) + bn6 + lrelu, lin2 (WMMA) + bias + bn7 + lrelu ---
  k_gemm4<<<dim3(8, 1, 1), 128, 0, stream>>>(gh, lin1h, L1, 32, 512, 512, 0, 0, 0,
                                             512, 512);
  k_epilogue<<<(B * 512 + 255) / 256, 256, 0, stream>>>(
      L1, nullptr, b6s, b6h, nullptr, 512, (long)B * 512, 2, nullptr, g1h, 512);
  k_gemm4<<<dim3(4, 1, 1), 128, 0, stream>>>(g1h, lin2h, L2, 32, 256, 512, 0, 0,
                                             0, 512, 512);
  k_epilogue<<<(B * 256 + 255) / 256, 256, 0, stream>>>(
      L2, F(IX_LIN2B), b7s, b7h, nullptr, 256, (long)B * 256, 2, g2, nullptr, 1);
  k_lin3<<<(B * 40 + 255) / 256, 256, 0, stream>>>(g2, F(IX_LIN3), F(IX_LIN3B),
                                                   (float*)d_out);
}
